// DepthScaleAwareDeformconv_12799002542776
// MI455X (gfx1250) — matrix operand order repaired and checked by reading, not compiled, not run on hardware
//
#include <hip/hip_runtime.h>
#include <hip/hip_bf16.h>

#define HH 96
#define WW 320
#define BB 4
#define HWSZ (HH*WW)

typedef __attribute__((ext_vector_type(16))) __bf16          v16bf;
typedef __attribute__((ext_vector_type(8)))  float           v8f;
typedef __attribute__((ext_vector_type(8)))  __bf16          v8bf;
typedef __attribute__((ext_vector_type(4)))  unsigned int    u32x4;

// Hardware RNE f32->bf16 (v_cvt_pk_bf16_f32 on gfx1250).
static __device__ __forceinline__ __bf16 f2bf(float f) { return (__bf16)f; }

// A-fragment (16x32 bf16): lane m = lane&15, half = lane>>4.
// element j in 0..15 -> K = chunkBase + half*16 + j        (contiguous 32B)
static __device__ __forceinline__ v16bf load_afrag(const __bf16* __restrict__ row,
                                                   int chunkBase, int half) {
    v8bf lo = *(const v8bf*)(row + chunkBase + half*16);
    v8bf hi = *(const v8bf*)(row + chunkBase + half*16 + 8);
    v16bf v;
#pragma unroll
    for (int i = 0; i < 8; i++) { v[i] = lo[i]; v[i+8] = hi[i]; }
    return v;
}

static __device__ __forceinline__ float actf(float v, int act) {
    if (act == 1) return fmaxf(v, 0.f);
    if (act == 2) return 1.f / (1.f + expf(-v));
    return v;
}

// ---------------------------------------------------------------- tiny MLP g
__global__ void g_kernel(const float* __restrict__ sb, const float* __restrict__ w1,
                         const float* __restrict__ b1, const float* __restrict__ w2,
                         const float* __restrict__ b2, float* __restrict__ g)
{
    int t = threadIdx.x;
    if (t >= 128) return;
    int b = t >> 5, o = t & 31;
    float s0 = sb[b*5+4] - sb[b*5+2];
    float s1 = sb[b*5+3] - sb[b*5+1];
    float acc = b2[o];
#pragma unroll 4
    for (int j = 0; j < 64; j++) {
        float h = fmaxf(s0*w1[j*2] + s1*w1[j*2+1] + b1[j], 0.f);
        acc += w2[o*64+j] * h;
    }
    g[b*32+o] = acc;
}

// --------------------------------------------- weight convert f32 -> bf16 (+pad)
__global__ void cvt_bf16(const float* __restrict__ src, __bf16* __restrict__ dst,
                         int rows_src, int cols, int total)
{
    int i = blockIdx.x*256 + threadIdx.x;
    if (i >= total) return;
    int row = i / cols;
    dst[i] = (row < rows_src) ? f2bf(src[i]) : (__bf16)0.f;
}

// ---------------------------------------------------- 3x3 conv, implicit GEMM
// M = Cout (MT tiles of 16), N = 64 pixels/block (16 per wave), K = 64*9 = 576.
// GSPLIT>0: channels [0,GSPLIT) are per-batch constants gvec (concat input).
template<int MT, int GSPLIT, int ACT>
__global__ __launch_bounds__(128) void conv3x3_wmma(
    const float* __restrict__ in, const float* __restrict__ gvec,
    const __bf16* __restrict__ wbf, const float* __restrict__ bias,
    float* __restrict__ out, int Cout)
{
    __shared__ __bf16 smem[64*3*68];
    const int strip = blockIdx.x;
    const int b  = strip / (HH*5);
    const int r0 = strip % (HH*5);
    const int y  = r0 / 5;
    const int x0 = (r0 % 5) * 64;
    const int tid  = threadIdx.x;
    const int inCh = 64 - GSPLIT;

    // Stage 64ch x 3rows x 66cols as bf16, SAME zero padding applied here.
    for (int e = tid; e < 64*3*66; e += 128) {
        int c   = e / 198;
        int rem = e - c*198;
        int row = rem / 66;
        int col = rem - row*66;
        int yy = y + row - 1;
        int xx = x0 + col - 1;
        float v = 0.f;
        if (yy >= 0 && yy < HH && xx >= 0 && xx < WW) {
            if (GSPLIT > 0 && c < GSPLIT) v = gvec[b*32 + c];
            else                          v = in[((size_t)(b*inCh + (c - GSPLIT))*HH + yy)*WW + xx];
        }
        smem[(c*3 + row)*68 + col] = f2bf(v);
    }
    __syncthreads();

    const int wave = tid >> 5, lane = tid & 31;
    const int half = lane >> 4, nm = lane & 15;
    const int pbase = wave*16 + nm;                // pixel column in strip

    const __bf16* rowp[MT];
#pragma unroll
    for (int mt = 0; mt < MT; mt++) {
        rowp[mt] = wbf + (size_t)(mt*16 + nm)*576;
        __builtin_prefetch(rowp[mt], 0, 0);        // global_prefetch_b8 (L2-warm weights)
    }

    v8f acc[MT];
#pragma unroll
    for (int mt = 0; mt < MT; mt++)
#pragma unroll
        for (int r = 0; r < 8; r++) acc[mt][r] = 0.f;

#pragma unroll
    for (int chunk = 0; chunk < 18; chunk++) {
        v16bf bf;                        // B frag: K = chunk*32 + half*16 + j, N = nm
#pragma unroll
        for (int j = 0; j < 16; j++) {
            // Both lane-halves' LDS offsets are compile-time; select with 1 cndmask.
            const int kkL = chunk*32 + j,        kkH = kkL + 16;
            const int cL  = kkL/9, tL = kkL-9*cL, cH = kkH/9, tH = kkH-9*cH;
            const int offL = (cL*3 + tL/3)*68 + (tL - (tL/3)*3);
            const int offH = (cH*3 + tH/3)*68 + (tH - (tH/3)*3);
            int off = offL + (half ? (offH - offL) : 0);
            bf[j] = smem[off + pbase];
        }
#pragma unroll
        for (int mt = 0; mt < MT; mt++) {
            v16bf af = load_afrag(rowp[mt], chunk*32, half);
            acc[mt] = __builtin_amdgcn_wmma_f32_16x16x32_bf16(
                false, af, false, bf, (short)0, acc[mt], false, false);
        }
    }

    // Store: immediate-offset form, base hoisted per (mt, half).
    const int xp = x0 + pbase;
#pragma unroll
    for (int mt = 0; mt < MT; mt++) {
        float* ob = out + ((size_t)(b*Cout + mt*16 + half*8)*HH + y)*WW + xp;
#pragma unroll
        for (int r = 0; r < 8; r++) {
            int o = mt*16 + half*8 + r;            // D: M = r + 8*half
            if (o < Cout)
                ob[(size_t)r*HWSZ] = actf(acc[mt][r] + bias[o], ACT);
        }
    }
}

// ---------------------------------------------------- 1x1 conv (Cin=32, K=32)
template<int MT, int ACT>
__global__ __launch_bounds__(128) void conv1x1_wmma(
    const float* __restrict__ in, const __bf16* __restrict__ wbf,
    const float* __restrict__ bias, float* __restrict__ out, int Cout)
{
    const int tid = threadIdx.x;
    const int wave = tid >> 5, lane = tid & 31, half = lane >> 4, nm = lane & 15;
    const int q = blockIdx.x*64 + wave*16 + nm;
    const int b = q / HWSZ, p = q - b*HWSZ;

    // Base hoisted: j*HWSZ folds into the 24-bit instruction offset.
    const float* pin = in + (size_t)b*32*HWSZ + (half ? 16*HWSZ : 0) + p;
    v16bf bf;
#pragma unroll
    for (int j = 0; j < 16; j++) bf[j] = f2bf(pin[(size_t)j*HWSZ]);

    v8f acc[MT];
#pragma unroll
    for (int mt = 0; mt < MT; mt++) {
        v8f z;
#pragma unroll
        for (int r = 0; r < 8; r++) z[r] = 0.f;
        v16bf af = load_afrag(wbf + (size_t)(mt*16 + nm)*32, 0, half);
        acc[mt] = __builtin_amdgcn_wmma_f32_16x16x32_bf16(
            false, af, false, bf, (short)0, z, false, false);
    }
#pragma unroll
    for (int mt = 0; mt < MT; mt++) {
        float* ob = out + (size_t)(b*Cout + mt*16 + half*8)*HWSZ + p;
#pragma unroll
        for (int r = 0; r < 8; r++) {
            int o = mt*16 + half*8 + r;
            if (o < Cout) ob[(size_t)r*HWSZ] = actf(acc[mt][r] + bias[o], ACT);
        }
    }
}

// ---------------------------------------------------- deformable conv via WMMA
// out[b,o,p] = bias[o] + sum_{c,k} bilinear(x_bc, p+tap_k+off) * mask * W[o,c,k]
// GEMM: M=64 (4 tiles), N=16 px/wave, K=576 (c*9+k), bf16 in, f32 accum.
// Tap table in LDS: 32B/record = {u32 byteoff0..3, f32 w0..3}, 2x ds_load_b128.
// Gathers use SADDR form: uniform base + zext 32-bit byte offset.
__global__ __launch_bounds__(128) void deform_wmma(
    const float* __restrict__ x, const float* __restrict__ offs,
    const float* __restrict__ mask, const __bf16* __restrict__ wbf,
    const float* __restrict__ bias, float* __restrict__ out)
{
    __shared__ unsigned int scombo[576*8];         // 64 px * 9 taps * (4 byteoff + 4 w)
    const int strip = blockIdx.x;
    const int b  = strip / (HH*5);
    const int r0 = strip % (HH*5);
    const int y  = r0 / 5;
    const int x0 = (r0 % 5) * 64;
    const int tid = threadIdx.x;
    const int wave = tid >> 5, lane = tid & 31, half = lane >> 4, nm = lane & 15;

    // Channel-independent bilinear taps: 64 pixels x 9 taps -> 4 byte-offsets + 4 weights.
    for (int e = tid; e < 64*9; e += 128) {
        int pl = e / 9;
        int k  = e - pl*9;
        int xpix = x0 + pl;
        size_t pix = (size_t)y*WW + xpix;
        float dy = offs[((size_t)(b*18 + 2*k    ))*HWSZ + pix];
        float dx = offs[((size_t)(b*18 + 2*k + 1))*HWSZ + pix];
        float m  = mask[((size_t)(b*9  + k     ))*HWSZ + pix];
        float ys = (float)y    + (float)(k/3 - 1) + dy;
        float xs = (float)xpix + (float)(k%3 - 1) + dx;
        float y0f = floorf(ys), x0f = floorf(xs);
        int iy0 = (int)y0f, ix0 = (int)x0f;
        float wy = ys - y0f, wx = xs - x0f;
        float w00 = (1.f-wy)*(1.f-wx)*m, w01 = (1.f-wy)*wx*m;
        float w10 = wy*(1.f-wx)*m,       w11 = wy*wx*m;
        int yy0 = min(max(iy0,   0), HH-1), yy1 = min(max(iy0+1, 0), HH-1);
        int xx0 = min(max(ix0,   0), WW-1), xx1 = min(max(ix0+1, 0), WW-1);
        bool vy0 = (iy0   >= 0) && (iy0   < HH);
        bool vy1 = (iy0+1 >= 0) && (iy0+1 < HH);
        bool vx0 = (ix0   >= 0) && (ix0   < WW);
        bool vx1 = (ix0+1 >= 0) && (ix0+1 < WW);
        int base = e*8;
        scombo[base+0] = (unsigned)(yy0*WW + xx0)*4u;  // pre-scaled byte offsets
        scombo[base+1] = (unsigned)(yy0*WW + xx1)*4u;
        scombo[base+2] = (unsigned)(yy1*WW + xx0)*4u;
        scombo[base+3] = (unsigned)(yy1*WW + xx1)*4u;
        scombo[base+4] = __builtin_bit_cast(unsigned, (vy0 && vx0) ? w00 : 0.f);
        scombo[base+5] = __builtin_bit_cast(unsigned, (vy0 && vx1) ? w01 : 0.f);
        scombo[base+6] = __builtin_bit_cast(unsigned, (vy1 && vx0) ? w10 : 0.f);
        scombo[base+7] = __builtin_bit_cast(unsigned, (vy1 && vx1) ? w11 : 0.f);
    }
    __syncthreads();

    const __bf16* rowp[4];
#pragma unroll
    for (int mt = 0; mt < 4; mt++) {
        rowp[mt] = wbf + (size_t)(mt*16 + nm)*576;
        __builtin_prefetch(rowp[mt], 0, 0);
    }

    v8f acc[4];
#pragma unroll
    for (int mt = 0; mt < 4; mt++)
#pragma unroll
        for (int r = 0; r < 8; r++) acc[mt][r] = 0.f;

    const char* xbb = (const char*)(x + (size_t)b*64*HWSZ);  // uniform base (saddr)
    const int pixrec = (wave*16 + nm)*9*8;         // lane's pixel record base (u32s)

#pragma unroll
    for (int chunk = 0; chunk < 18; chunk++) {
        v16bf bf;
#pragma unroll
        for (int j = 0; j < 16; j++) {
            // Both lane-halves' (tap record, channel displacement) are compile-time.
            const int kkL = chunk*32 + j,          kkH = kkL + 16;
            const int cL  = kkL/9, kL = kkL-9*cL,  cH = kkH/9, kH = kkH-9*cH;
            const int      recL = kL*8,            recH = kH*8;           // u32s
            const unsigned chL  = (unsigned)cL*HWSZ*4u,
                           chH  = (unsigned)cH*HWSZ*4u;                   // bytes
            int      rec = recL + (half ? (recH - recL) : 0);
            unsigned chB = chL  + (half ? (chH  - chL ) : 0u);
            const unsigned int* cb = scombo + pixrec + rec;
            u32x4  i4 = *(const u32x4*)(cb);
            float4 w4 = __builtin_bit_cast(float4, *(const u32x4*)(cb + 4));
            // unsigned 32-bit offsets -> zext GEP -> global_load SADDR form
            float v = w4.x * *(const float*)(xbb + (chB + i4[0]))
                    + w4.y * *(const float*)(xbb + (chB + i4[1]))
                    + w4.z * *(const float*)(xbb + (chB + i4[2]))
                    + w4.w * *(const float*)(xbb + (chB + i4[3]));
            bf[j] = f2bf(v);
        }
#pragma unroll
        for (int mt = 0; mt < 4; mt++) {
            v16bf af = load_afrag(rowp[mt], chunk*32, half);
            acc[mt] = __builtin_amdgcn_wmma_f32_16x16x32_bf16(
                false, af, false, bf, (short)0, acc[mt], false, false);
        }
    }

    const int xpix = x0 + wave*16 + nm;
#pragma unroll
    for (int mt = 0; mt < 4; mt++) {
        float* ob = out + ((size_t)(b*64 + mt*16 + half*8)*HH + y)*WW + xpix;
#pragma unroll
        for (int r = 0; r < 8; r++) {
            int o = mt*16 + half*8 + r;
            ob[(size_t)r*HWSZ] = acc[mt][r] + bias[o];
        }
    }
}

// ------------------------------------------------------------------ launcher
extern "C" void kernel_launch(void* const* d_in, const int* in_sizes, int n_in,
                              void* d_out, int out_size, void* d_ws, size_t ws_size,
                              hipStream_t stream) {
    (void)in_sizes; (void)n_in; (void)out_size; (void)ws_size;
    const float* x        = (const float*)d_in[0];
    const float* sb       = (const float*)d_in[1];
    const float* glob_w1  = (const float*)d_in[2];
    const float* glob_b1  = (const float*)d_in[3];
    const float* glob_w2  = (const float*)d_in[4];
    const float* glob_b2  = (const float*)d_in[5];
    const float* loc_w1   = (const float*)d_in[6];
    const float* loc_b1   = (const float*)d_in[7];
    const float* loc_w2   = (const float*)d_in[8];
    const float* loc_b2   = (const float*)d_in[9];
    const float* fuse_w1  = (const float*)d_in[10];
    const float* fuse_b1  = (const float*)d_in[11];
    const float* fuse_w2  = (const float*)d_in[12];
    const float* fuse_b2  = (const float*)d_in[13];
    const float* mask_w   = (const float*)d_in[14];
    const float* mask_b   = (const float*)d_in[15];
    const float* conv_w   = (const float*)d_in[16];
    const float* conv_b   = (const float*)d_in[17];

    char* ws = (char*)d_ws;
    const size_t szPlane32 = (size_t)BB*32*HWSZ*sizeof(float);   // 15,728,640
    float* gbuf    = (float*)(ws);                                // 512 B
    float* loc1    = (float*)(ws + 1024);
    float* lbuf    = (float*)(ws + 1024 +   szPlane32);
    float* fbuf    = (float*)(ws + 1024 + 2*szPlane32);
    float* offbuf  = (float*)(ws + 1024 + 3*szPlane32);           // B*18*HW f32
    float* maskbuf = (float*)(ws + 1024 + 3*szPlane32 + (size_t)BB*18*HWSZ*sizeof(float));
    char*  wbase   = (char*)maskbuf + (size_t)BB*9*HWSZ*sizeof(float);
    __bf16* wbLoc1  = (__bf16*)(wbase);                            // 32x576
    __bf16* wbFuse1 = wbLoc1  + 32*576;                            // 32x576
    __bf16* wbConv  = wbFuse1 + 32*576;                            // 64x576
    __bf16* wbLoc2  = wbConv  + 64*576;                            // 32x32
    __bf16* wbFuse2 = wbLoc2  + 32*32;                             // 32x32 (18 real)
    __bf16* wbMask  = wbFuse2 + 32*32;                             // 16x576 (9 real)

    // Weight conversion to bf16 (padded Cout rows zeroed).
    cvt_bf16<<<(32*576 + 255)/256, 256, 0, stream>>>(loc_w1,  wbLoc1,  32, 576, 32*576);
    cvt_bf16<<<(32*576 + 255)/256, 256, 0, stream>>>(fuse_w1, wbFuse1, 32, 576, 32*576);
    cvt_bf16<<<(64*576 + 255)/256, 256, 0, stream>>>(conv_w,  wbConv,  64, 576, 64*576);
    cvt_bf16<<<(32*32  + 255)/256, 256, 0, stream>>>(loc_w2,  wbLoc2,  32,  32, 32*32);
    cvt_bf16<<<(32*32  + 255)/256, 256, 0, stream>>>(fuse_w2, wbFuse2, 18,  32, 32*32);
    cvt_bf16<<<(16*576 + 255)/256, 256, 0, stream>>>(mask_w,  wbMask,   9, 576, 16*576);

    g_kernel<<<1, 128, 0, stream>>>(sb, glob_w1, glob_b1, glob_w2, glob_b2, gbuf);

    const int strips = BB * HH * (WW/64);   // 1920 blocks, 4 waves each
    // loc1 = relu(conv3x3(x, loc_w1))
    conv3x3_wmma<2, 0, 1><<<strips, 128, 0, stream>>>(x,    gbuf, wbLoc1,  loc_b1,  loc1,   32);
    // l = conv1x1(loc1, loc_w2)
    conv1x1_wmma<2, 0>   <<<strips, 128, 0, stream>>>(loc1,       wbLoc2,  loc_b2,  lbuf,   32);
    // f = relu(conv3x3(concat[g, l], fuse_w1))
    conv3x3_wmma<2, 32, 1><<<strips, 128, 0, stream>>>(lbuf, gbuf, wbFuse1, fuse_b1, fbuf,   32);
    // offset = conv1x1(f, fuse_w2)
    conv1x1_wmma<2, 0>   <<<strips, 128, 0, stream>>>(fbuf,       wbFuse2, fuse_b2, offbuf, 18);
    // mask = sigmoid(conv3x3(x, mask_w))
    conv3x3_wmma<1, 0, 2><<<strips, 128, 0, stream>>>(x,    gbuf, wbMask,  mask_b,  maskbuf, 9);
    // out = deform_conv(x, offset, mask, conv_w) + conv_b
    deform_wmma<<<strips, 128, 0, stream>>>(x, offbuf, maskbuf, wbConv, conv_b, (float*)d_out);
}
